// Kernel_69243462746802
// MI455X (gfx1250) — compile-verified, matrix operand order checked by reading, not run on hardware
//
#include <hip/hip_runtime.h>
#include <hip/hip_bf16.h>

typedef __attribute__((ext_vector_type(16))) _Float16 v16h;
typedef __attribute__((ext_vector_type(8)))  float    v8f;

#define B_  2
#define H_  8
#define N_  2048
#define P_  64
#define M_  4
#define D_  256
#define NX_ (2 * N_)      // 4096 rows of X = concat(Q,K)
#define MD_ (M_ * D_)     // 1024 sketch rows per (b,h)

// ---------------------------------------------------------------------------
// Kernel 1: gather the sketch rows out of X = concat(Q,K) and pack as f16.
// Wpack layout: [bh][md][p], row-major, 64 halves (128 B) per row.
// One block per (md, bh) row, 64 threads = one element each.
// ---------------------------------------------------------------------------
__global__ void __launch_bounds__(64)
gather_pack(const float* __restrict__ Q,
            const float* __restrict__ K,
            const int*   __restrict__ sk,     // (B, M, D) int32, values in [0, 2N)
            _Float16*    __restrict__ W) {
  const int md = blockIdx.x;                  // 0..1023  (= m*256 + d)
  const int bh = blockIdx.y;                  // 0..15    (= b*8 + h)
  const int b  = bh >> 3;
  const int p  = threadIdx.x;                 // 0..63
  const int j  = sk[b * MD_ + md];            // row of X to gather (same for all h)
  const float* src = (j < N_)
      ? (Q + ((size_t)bh * N_ + j) * P_)
      : (K + ((size_t)bh * N_ + (j - N_)) * P_);
  W[((size_t)bh * MD_ + md) * P_ + p] = (_Float16)src[p];
}

// ---------------------------------------------------------------------------
// Kernel 2: G = X @ W^T via v_wmma_f32_16x16x32_f16, then
//           AS[n,d] = sum_m sign[m,d] * exp(G[n, m*256+d]).
// Grid: x = 4096/32 n-tiles, y = 16 (b,h).  Block: 256 threads = 8 waves.
// Wave (sub,g): sub = 16-row n-subtile, g = exclusive 64-column d-group, so the
// m-reduction stays in registers and every output element is written once.
// ---------------------------------------------------------------------------
__global__ void __launch_bounds__(256)
sketch_attn(const float*    __restrict__ Q,
            const float*    __restrict__ K,
            const _Float16* __restrict__ W,     // packed f16 sketch rows
            const float*    __restrict__ sign,  // (M, D) f32 of +/-1
            float*          __restrict__ out) { // (B, H, 2N, D) f32
  const int bh   = blockIdx.y;
  const int n0   = blockIdx.x * 32;
  const int wave = threadIdx.x >> 5;
  const int lane = threadIdx.x & 31;
  const int sub  = wave & 1;          // which 16-row n-subtile
  const int g    = wave >> 1;         // which 64-column d-group
  const int lrow = lane & 15;
  const int lhi  = lane >> 4;         // 0 or 1: selects K sub-range per ISA layout

  // ----- A fragments (16x32 f16 each), loaded once, reused for all 16 tiles.
  // 16-bit A layout: lanes 0-15 hold K in {kb..kb+7} U {kb+16..kb+23}, kb = lhi*8.
  // Subtiles are 16-row aligned so the Q/K pointer select is wave-uniform
  // (EXEC stays all-1s around the WMMAs).
  const int nrow = n0 + sub * 16 + lrow;
  const float* xrow = (nrow < N_)
      ? (Q + ((size_t)bh * N_ + nrow) * P_)
      : (K + ((size_t)bh * N_ + (nrow - N_)) * P_);
  const int kb = lhi * 8;

  v16h a0, a1;   // K chunk 0: p in [0,32), chunk 1: p in [32,64)
#pragma unroll
  for (int i = 0; i < 8; ++i) {
    a0[i]     = (_Float16)xrow[kb + i];
    a0[8 + i] = (_Float16)xrow[kb + 16 + i];
    a1[i]     = (_Float16)xrow[32 + kb + i];
    a1[8 + i] = (_Float16)xrow[32 + kb + 16 + i];
  }

  const _Float16* wbh  = W + (size_t)bh * MD_ * P_;
  const int       dcol0 = g * 64 + lrow;

#pragma unroll
  for (int dt = 0; dt < 4; ++dt) {
    const int dbase = g * 64 + dt * 16;
    const int dcol  = dbase + lrow;          // this lane's output column

    v8f res = {0.f, 0.f, 0.f, 0.f, 0.f, 0.f, 0.f, 0.f};

#pragma unroll
    for (int m = 0; m < M_; ++m) {
      // B matrix is 32x16 (KxN): lane's column n = lrow, K sub-ranges as in A.
      const _Float16* wr = wbh + ((size_t)(m * D_ + dcol)) * P_;

      // Prefetch next m's row for this lane (global_prefetch_b8).
      if (m + 1 < M_)
        __builtin_prefetch(wbh + ((size_t)((m + 1) * D_ + dcol)) * P_, 0, 1);

      v16h b0, b1;
#pragma unroll
      for (int i = 0; i < 8; ++i) {
        b0[i]     = wr[kb + i];
        b0[8 + i] = wr[kb + 16 + i];
        b1[i]     = wr[32 + kb + i];
        b1[8 + i] = wr[32 + kb + 16 + i];
      }

      v8f gacc = {0.f, 0.f, 0.f, 0.f, 0.f, 0.f, 0.f, 0.f};
      gacc = __builtin_amdgcn_wmma_f32_16x16x32_f16(
          false, a0, false, b0, (short)0, gacc, false, false);
      gacc = __builtin_amdgcn_wmma_f32_16x16x32_f16(
          false, a1, false, b1, (short)0, gacc, false, false);

      const float sg = sign[m * D_ + dcol];
#pragma unroll
      for (int v = 0; v < 8; ++v)
        res[v] += sg * __expf(gacc[v]);     // v_exp_f32 on the TRANS pipe
    }

    // 16x16 f32 C/D layout: lane {0-15,16-31} -> rows {v, 8+v}, col = lrow.
    const size_t rbase = (size_t)bh * NX_ + (n0 + sub * 16 + lhi * 8);
#pragma unroll
    for (int v = 0; v < 8; ++v)
      out[(rbase + v) * D_ + dcol] = res[v];
  }
  (void)dcol0;
}

extern "C" void kernel_launch(void* const* d_in, const int* in_sizes, int n_in,
                              void* d_out, int out_size, void* d_ws, size_t ws_size,
                              hipStream_t stream) {
  const float* Q    = (const float*)d_in[0];   // (B,H,N,P) f32
  const float* K    = (const float*)d_in[1];   // (B,H,N,P) f32
  const int*   sk   = (const int*)d_in[2];     // (B,M,D) int32
  const float* sign = (const float*)d_in[3];   // (M,D) f32
  float*       out  = (float*)d_out;           // (B,H,2N,D) f32

  _Float16* W = (_Float16*)d_ws;               // 16*1024*64*2 B = 2 MB scratch

  dim3 g1(MD_, B_ * H_);
  gather_pack<<<g1, 64, 0, stream>>>(Q, K, sk, W);

  dim3 g2(NX_ / 32, B_ * H_);
  sketch_attn<<<g2, 256, 0, stream>>>(Q, K, W, sign, out);
}
